// Block_55551107006967
// MI455X (gfx1250) — matrix operand order repaired and checked by reading, not compiled, not run on hardware
//
#include <hip/hip_runtime.h>

typedef __bf16 bf16_t;
typedef __attribute__((ext_vector_type(16))) __bf16 v16bf;
typedef __attribute__((ext_vector_type(8)))  __bf16 v8bf;
typedef __attribute__((ext_vector_type(8)))  float  v8f;
typedef __attribute__((ext_vector_type(4)))  unsigned u32x4;
typedef __attribute__((ext_vector_type(8)))  int  i32x8;
typedef __attribute__((ext_vector_type(4)))  int  i32x4;

#define T_SEQ  2048
#define C_DIM  2048
#define N_HEAD 16
#define N_KV   8
#define HS     128
#define FFN    5632
#define QKV_D  ((N_HEAD + 2 * N_KV) * HS)   // 4096

#if defined(__has_builtin)
#if __has_builtin(__builtin_amdgcn_tensor_load_to_lds) && __has_builtin(__builtin_amdgcn_s_wait_tensorcnt)
#define USE_TDM 1
#endif
#endif
#ifndef USE_TDM
#define USE_TDM 0
#endif

__device__ __forceinline__ v16bf cat8(v8bf a, v8bf b) {
  return __builtin_shufflevector(a, b, 0,1,2,3,4,5,6,7,8,9,10,11,12,13,14,15);
}

__device__ __forceinline__ float half_max16(float v) {
  v = fmaxf(v, __shfl_xor(v, 1, 32));
  v = fmaxf(v, __shfl_xor(v, 2, 32));
  v = fmaxf(v, __shfl_xor(v, 4, 32));
  v = fmaxf(v, __shfl_xor(v, 8, 32));
  return v;
}
__device__ __forceinline__ float half_sum16(float v) {
  v += __shfl_xor(v, 1, 32);
  v += __shfl_xor(v, 2, 32);
  v += __shfl_xor(v, 4, 32);
  v += __shfl_xor(v, 8, 32);
  return v;
}

#if USE_TDM
// Issue one TDM 2D tile load: rows=128 x 32 bf16 elements, row stride = stride_elems,
// LDS destination padded 4 DWORDs every 16 DWORDs (=> 40-element LDS row stride).
__device__ __forceinline__ void tdm_issue_tile(unsigned lds_off, const bf16_t* gptr,
                                               int stride_elems) {
  unsigned long long ga = (unsigned long long)(size_t)gptr;
  u32x4 g0;
  g0.x = 1u;                                            // count=1 (valid user D#)
  g0.y = lds_off;                                       // LDS byte address
  g0.z = (unsigned)ga;                                  // global addr [31:0]
  g0.w = (unsigned)((ga >> 32) & 0x01FFFFFFull) | (2u << 30);  // addr[56:32] | type=2
  const unsigned td0 = 1u << 20, td1 = 1u << 20;        // huge tensor dims (tiles in-bounds)
  i32x8 g1;
  g1[0] = (int)((1u << 16)      // data_size = 2 bytes
              | (1u << 20)      // pad_enable
              | (3u << 22)      // pad_interval: 16 DWORDs
              | (3u << 25));    // pad_amount: 4 DWORDs
  g1[1] = (int)((td0 & 0xFFFFu) << 16);                               // dim0.lo16
  g1[2] = (int)(((td0 >> 16) & 0xFFFFu) | ((td1 & 0xFFFFu) << 16));   // dim0.hi|dim1.lo
  g1[3] = (int)(((td1 >> 16) & 0xFFFFu) | (32u << 16));               // dim1.hi|tile_dim0=32
  g1[4] = (int)(128u);                                                // tile_dim1=128, tile_dim2=0
  g1[5] = stride_elems;                                               // dim0_stride lo32
  g1[6] = 0;
  g1[7] = 0;
  i32x4 z4 = {0, 0, 0, 0};
#if __clang_major__ >= 23
  i32x8 z8 = {0, 0, 0, 0, 0, 0, 0, 0};
  __builtin_amdgcn_tensor_load_to_lds(g0, g1, z4, z4, z8, 0);
#else
  __builtin_amdgcn_tensor_load_to_lds(g0, g1, z4, z4, 0);
#endif
}
#endif

// ---------------- elementwise kernels ----------------

__global__ __launch_bounds__(256) void f32_to_bf16_kernel(const float* __restrict__ s,
                                                          bf16_t* __restrict__ d, int n) {
  for (int i = blockIdx.x * 256 + threadIdx.x; i < n; i += gridDim.x * 256)
    d[i] = (bf16_t)s[i];
}

__global__ __launch_bounds__(256) void add_f32_kernel(const float* __restrict__ a,
                                                      const float* __restrict__ b,
                                                      float* __restrict__ o, int n) {
  int i = blockIdx.x * 256 + threadIdx.x;
  if (i < n) o[i] = a[i] + b[i];
}

__global__ __launch_bounds__(256) void silu_mul_kernel(const float* __restrict__ h1,
                                                       const float* __restrict__ h2,
                                                       bf16_t* __restrict__ o, int n) {
  for (int i = blockIdx.x * 256 + threadIdx.x; i < n; i += gridDim.x * 256) {
    float a = h1[i];
    float s = a / (1.0f + __expf(-a));
    o[i] = (bf16_t)(s * h2[i]);
  }
}

// ---------------- RMSNorm: one block per row ----------------

__global__ __launch_bounds__(256) void rmsnorm_kernel(const float* __restrict__ x,
                                                      const float* __restrict__ w,
                                                      bf16_t* __restrict__ out) {
  __shared__ float red[256];
  const int row = blockIdx.x;
  const float* xr = x + (size_t)row * C_DIM;
  float ss = 0.f;
  for (int c = threadIdx.x; c < C_DIM; c += 256) { float v = xr[c]; ss += v * v; }
  red[threadIdx.x] = ss;
  __syncthreads();
  for (int s = 128; s > 0; s >>= 1) {
    if (threadIdx.x < s) red[threadIdx.x] += red[threadIdx.x + s];
    __syncthreads();
  }
  const float inv = rsqrtf(red[0] * (1.0f / C_DIM) + 1e-5f);
  for (int c = threadIdx.x; c < C_DIM; c += 256)
    out[(size_t)row * C_DIM + c] = (bf16_t)(xr[c] * inv * w[c]);
}

// ---------------- NT bf16 GEMM: C[M,N] = A[M,K] * B[N,K]^T ----------------
// A, B bf16 row-major (K contiguous); C f32 row-major. M,N mult of 128; K mult of 32.
// TDM path: double-buffered LDS tiles, wave0 DMAs next tile while all waves compute.

__global__ __launch_bounds__(256) void gemm_nt_bf16_kernel(const bf16_t* __restrict__ A,
                                                           const bf16_t* __restrict__ B,
                                                           float* __restrict__ C,
                                                           int M, int N, int K) {
  __shared__ bf16_t As[2][128][40];   // 80B row stride: 16B-aligned, conflict-free
  __shared__ bf16_t Bs[2][128][40];
  const int tid  = threadIdx.x;
  const int wave = tid >> 5, lane = tid & 31;
  const int lo16 = lane & 15, hi = lane >> 4;
  const int wrow = wave >> 2, wcol = wave & 3;       // 2 x 4 waves
  const int m0 = blockIdx.y * 128, n0 = blockIdx.x * 128;

  v8f acc[4][2];
#pragma unroll
  for (int mi = 0; mi < 4; ++mi)
#pragma unroll
    for (int ni = 0; ni < 2; ++ni)
      acc[mi][ni] = v8f{0.f,0.f,0.f,0.f,0.f,0.f,0.f,0.f};

#if USE_TDM
  const bf16_t* Abase = A + (size_t)m0 * K;
  const bf16_t* Bbase = B + (size_t)n0 * K;
  const unsigned ldsA0 = (unsigned)(size_t)(void*)&As[0][0][0];
  const unsigned ldsA1 = (unsigned)(size_t)(void*)&As[1][0][0];
  const unsigned ldsB0 = (unsigned)(size_t)(void*)&Bs[0][0][0];
  const unsigned ldsB1 = (unsigned)(size_t)(void*)&Bs[1][0][0];
  if (wave == 0) {
    tdm_issue_tile(ldsA0, Abase, K);
    tdm_issue_tile(ldsB0, Bbase, K);
  }
  int cur = 0;
  for (int kb = 0; kb < K; kb += 32) {
    if (wave == 0) {
      if (kb + 32 < K) {               // DMA next tile into the other buffer
        tdm_issue_tile(cur ? ldsA0 : ldsA1, Abase + kb + 32, K);
        tdm_issue_tile(cur ? ldsB0 : ldsB1, Bbase + kb + 32, K);
        __builtin_amdgcn_s_wait_tensorcnt(2);   // current buffer's 2 DMAs retired (in-order)
      } else {
        __builtin_amdgcn_s_wait_tensorcnt(0);
      }
    }
    __syncthreads();

    v16bf af[4], bfr[2];
#pragma unroll
    for (int mi = 0; mi < 4; ++mi) {
      const bf16_t* p = &As[cur][wrow * 64 + mi * 16 + lo16][hi * 8];
      af[mi] = cat8(*(const v8bf*)p, *(const v8bf*)(p + 16));
    }
#pragma unroll
    for (int ni = 0; ni < 2; ++ni) {
      const bf16_t* p = &Bs[cur][wcol * 32 + ni * 16 + lo16][hi * 8];
      bfr[ni] = cat8(*(const v8bf*)p, *(const v8bf*)(p + 16));
    }
#pragma unroll
    for (int mi = 0; mi < 4; ++mi)
#pragma unroll
      for (int ni = 0; ni < 2; ++ni)
        acc[mi][ni] = __builtin_amdgcn_wmma_f32_16x16x32_bf16(
            false, af[mi], false, bfr[ni], (short)0, acc[mi][ni], false, false);
    __syncthreads();
    cur ^= 1;
  }
#else
  const int lr = tid >> 1;          // load row 0..127
  const int lc = (tid & 1) * 16;    // k half
  for (int kb = 0; kb < K; kb += 32) {
    const bf16_t* ag = A + (size_t)(m0 + lr) * K + kb + lc;
    const bf16_t* bg = B + (size_t)(n0 + lr) * K + kb + lc;
    v8bf a0 = *(const v8bf*)ag, a1 = *(const v8bf*)(ag + 8);
    v8bf b0 = *(const v8bf*)bg, b1 = *(const v8bf*)(bg + 8);
    if (kb + 32 < K) {               // prefetch next K tile (global_prefetch_b8)
      __builtin_prefetch(ag + 32, 0, 1);
      __builtin_prefetch(bg + 32, 0, 1);
    }
    __syncthreads();
    *(v8bf*)&As[0][lr][lc] = a0; *(v8bf*)&As[0][lr][lc + 8] = a1;
    *(v8bf*)&Bs[0][lr][lc] = b0; *(v8bf*)&Bs[0][lr][lc + 8] = b1;
    __syncthreads();

    v16bf af[4], bfr[2];
#pragma unroll
    for (int mi = 0; mi < 4; ++mi) {
      const bf16_t* p = &As[0][wrow * 64 + mi * 16 + lo16][hi * 8];
      af[mi] = cat8(*(const v8bf*)p, *(const v8bf*)(p + 16));
    }
#pragma unroll
    for (int ni = 0; ni < 2; ++ni) {
      const bf16_t* p = &Bs[0][wcol * 32 + ni * 16 + lo16][hi * 8];
      bfr[ni] = cat8(*(const v8bf*)p, *(const v8bf*)(p + 16));
    }
#pragma unroll
    for (int mi = 0; mi < 4; ++mi)
#pragma unroll
      for (int ni = 0; ni < 2; ++ni)
        acc[mi][ni] = __builtin_amdgcn_wmma_f32_16x16x32_bf16(
            false, af[mi], false, bfr[ni], (short)0, acc[mi][ni], false, false);
  }
#endif

#pragma unroll
  for (int mi = 0; mi < 4; ++mi)
#pragma unroll
    for (int ni = 0; ni < 2; ++ni)
#pragma unroll
      for (int r = 0; r < 8; ++r)
        C[(size_t)(m0 + wrow * 64 + mi * 16 + r + hi * 8) * N +
          (n0 + wcol * 32 + ni * 16 + lo16)] = acc[mi][ni][r];
}

// ---------------- RoPE + split QKV ----------------
// qkv f32 (T x 4096) -> q_bf (T x 2048), k_bf (T x 1024), v_t bf16 (1024 x T)

__global__ __launch_bounds__(256) void rope_split_kernel(const float* __restrict__ qkv,
                                                         const float* __restrict__ cosv,
                                                         const float* __restrict__ sinv,
                                                         bf16_t* __restrict__ qb,
                                                         bf16_t* __restrict__ kb,
                                                         bf16_t* __restrict__ vt) {
  const int t = blockIdx.x;
  const float* row = qkv + (size_t)t * QKV_D;
  for (int col = threadIdx.x; col < QKV_D; col += 256) {
    const int d = col & (HS - 1);
    const float v = row[col];
    if (col < (N_HEAD + N_KV) * HS) {  // q or k: apply rope
      const int base = col - d;
      const float rot = (d < HS / 2) ? -row[base + d + HS / 2] : row[base + d - HS / 2];
      const float o = v * cosv[t * HS + d] + rot * sinv[t * HS + d];
      if (col < N_HEAD * HS) qb[(size_t)t * (N_HEAD * HS) + col] = (bf16_t)o;
      else                   kb[(size_t)t * (N_KV * HS) + (col - N_HEAD * HS)] = (bf16_t)o;
    } else {                           // v: transpose store
      vt[(size_t)(col - (N_HEAD + N_KV) * HS) * T_SEQ + t] = (bf16_t)v;
    }
  }
}

// ---------------- Flash attention: one wave per (head, 16-query tile) ----------------

__global__ __launch_bounds__(256) void flash_attn_kernel(const bf16_t* __restrict__ qb,
                                                         const bf16_t* __restrict__ kbm,
                                                         const bf16_t* __restrict__ vt,
                                                         bf16_t* __restrict__ yb) {
  __shared__ bf16_t pstage[8][16][40];   // per-wave prob staging (D-layout -> A-layout)
  const int wave = threadIdx.x >> 5;
  const int lane = threadIdx.x & 31;
  const int lo16 = lane & 15, hi = lane >> 4;
  const int h  = blockIdx.y;
  const int kv = h >> 1;                 // GQA: 2 query heads per kv head
  const int t0 = (blockIdx.x * 8 + wave) * 16;
  const float scale = 0.08838834764831845f;  // 1/sqrt(128)

  // Q fragments (A-layout), loaded once
  v16bf qf[4];
  {
    const bf16_t* qrow = qb + (size_t)(t0 + lo16) * (N_HEAD * HS) + h * HS;
#pragma unroll
    for (int kk = 0; kk < 4; ++kk) {
      const int c0 = kk * 32 + hi * 8;
      qf[kk] = cat8(*(const v8bf*)(qrow + c0), *(const v8bf*)(qrow + c0 + 16));
    }
  }

  v8f acc[8];
#pragma unroll
  for (int n = 0; n < 8; ++n) acc[n] = v8f{0.f,0.f,0.f,0.f,0.f,0.f,0.f,0.f};
  float mrow[8], lrow[8];
#pragma unroll
  for (int r = 0; r < 8; ++r) { mrow[r] = -3.0e38f; lrow[r] = 0.f; }

  const int nk = (t0 + 16 + 31) & ~31;   // keys processed (padded; pads are masked)
  for (int kb = 0; kb < nk; kb += 32) {
    // scores: two 16x16 tiles covering 32 keys
    v8f s[2];
#pragma unroll
    for (int j = 0; j < 2; ++j) {
      v8f sj = v8f{0.f,0.f,0.f,0.f,0.f,0.f,0.f,0.f};
      const bf16_t* krow = kbm + (size_t)(kb + j * 16 + lo16) * (N_KV * HS) + kv * HS + hi * 8;
#pragma unroll
      for (int kk = 0; kk < 4; ++kk) {
        v16bf bfr = cat8(*(const v8bf*)(krow + kk * 32), *(const v8bf*)(krow + kk * 32 + 16));
        sj = __builtin_amdgcn_wmma_f32_16x16x32_bf16(false, qf[kk], false, bfr,
                                                     (short)0, sj, false, false);
      }
      s[j] = sj;
    }

    // causal mask + online softmax (rows live in VGPR index, cols in lanes)
    float alpha[8];
#pragma unroll
    for (int r = 0; r < 8; ++r) {
      const int q = t0 + r + hi * 8;
      float s0 = s[0][r] * scale; if (kb + lo16 > q)      s0 = -1.0e30f;
      float s1 = s[1][r] * scale; if (kb + 16 + lo16 > q) s1 = -1.0e30f;
      const float mc = half_max16(fmaxf(s0, s1));
      const float mn = fmaxf(mrow[r], mc);
      const float al = __expf(mrow[r] - mn);
      mrow[r] = mn;
      const float p0 = __expf(s0 - mn);
      const float p1 = __expf(s1 - mn);
      lrow[r] = lrow[r] * al + half_sum16(p0 + p1);
      alpha[r] = al;
      pstage[wave][r + hi * 8][lo16]      = (bf16_t)p0;
      pstage[wave][r + hi * 8][16 + lo16] = (bf16_t)p1;
    }
#pragma unroll
    for (int n = 0; n < 8; ++n)
#pragma unroll
      for (int r = 0; r < 8; ++r)
        acc[n][r] *= alpha[r];

    asm volatile("s_wait_dscnt 0" ::: "memory");   // LDS RAW within wave

    // probs A-fragment + 8 WMMAs of P x V^T
    const bf16_t* pr = &pstage[wave][lo16][hi * 8];
    v16bf pa = cat8(*(const v8bf*)pr, *(const v8bf*)(pr + 16));
#pragma unroll
    for (int n = 0; n < 8; ++n) {
      const bf16_t* vrow = vt + (size_t)(kv * HS + n * 16 + lo16) * T_SEQ + kb + hi * 8;
      v16bf vb = cat8(*(const v8bf*)vrow, *(const v8bf*)(vrow + 16));
      acc[n] = __builtin_amdgcn_wmma_f32_16x16x32_bf16(false, pa, false, vb,
                                                       (short)0, acc[n], false, false);
    }
  }

  // normalize and write y (bf16, feeds proj GEMM)
  float linv[8];
#pragma unroll
  for (int r = 0; r < 8; ++r) linv[r] = 1.0f / lrow[r];
#pragma unroll
  for (int n = 0; n < 8; ++n)
#pragma unroll
    for (int r = 0; r < 8; ++r)
      yb[(size_t)(t0 + r + hi * 8) * (N_HEAD * HS) + h * HS + n * 16 + lo16] =
          (bf16_t)(acc[n][r] * linv[r]);
}

// ---------------- orchestration ----------------

extern "C" void kernel_launch(void* const* d_in, const int* in_sizes, int n_in,
                              void* d_out, int out_size, void* d_ws, size_t ws_size,
                              hipStream_t stream) {
  (void)in_sizes; (void)n_in; (void)out_size; (void)ws_size;
  const float* x        = (const float*)d_in[0];
  const float* cosv     = (const float*)d_in[1];
  const float* sinv     = (const float*)d_in[2];
  /* d_in[3] token_idx unused */
  const float* norm1_w  = (const float*)d_in[4];
  const float* qkv_w    = (const float*)d_in[5];
  const float* proj_w   = (const float*)d_in[6];
  const float* norm2_w  = (const float*)d_in[7];
  const float* fc1_w    = (const float*)d_in[8];
  const float* fc2_w    = (const float*)d_in[9];
  const float* mlp_w    = (const float*)d_in[10];
  float* out = (float*)d_out;

  char* wsp = (char*)d_ws;
  auto alloc = [&](size_t bytes) -> char* {
    char* p = wsp;
    wsp += (bytes + 255) & ~(size_t)255;
    return p;
  };

  bf16_t* qkvw_bf = (bf16_t*)alloc((size_t)QKV_D * C_DIM * 2);
  bf16_t* projw_bf= (bf16_t*)alloc((size_t)C_DIM * (N_HEAD * HS) * 2);
  bf16_t* fc1w_bf = (bf16_t*)alloc((size_t)FFN * C_DIM * 2);
  bf16_t* fc2w_bf = (bf16_t*)alloc((size_t)FFN * C_DIM * 2);
  bf16_t* mlpw_bf = (bf16_t*)alloc((size_t)C_DIM * FFN * 2);
  bf16_t* xn_bf   = (bf16_t*)alloc((size_t)T_SEQ * C_DIM * 2);
  float*  qkv_f   = (float*) alloc((size_t)T_SEQ * QKV_D * 4);
  bf16_t* q_bf    = (bf16_t*)alloc((size_t)T_SEQ * N_HEAD * HS * 2);
  bf16_t* k_bf    = (bf16_t*)alloc((size_t)T_SEQ * N_KV * HS * 2);
  bf16_t* vt_bf   = (bf16_t*)alloc((size_t)N_KV * HS * T_SEQ * 2);
  bf16_t* y_bf    = (bf16_t*)alloc((size_t)T_SEQ * N_HEAD * HS * 2);
  float*  attnp_f = (float*) alloc((size_t)T_SEQ * C_DIM * 4);
  float*  x2_f    = (float*) alloc((size_t)T_SEQ * C_DIM * 4);
  bf16_t* xn2_bf  = (bf16_t*)alloc((size_t)T_SEQ * C_DIM * 2);
  float*  h1_f    = (float*) alloc((size_t)T_SEQ * FFN * 4);
  float*  h2_f    = (float*) alloc((size_t)T_SEQ * FFN * 4);
  bf16_t* h_bf    = (bf16_t*)alloc((size_t)T_SEQ * FFN * 2);
  float*  mlp_f   = (float*) alloc((size_t)T_SEQ * C_DIM * 4);

  // 1. weights -> bf16
  f32_to_bf16_kernel<<<2048, 256, 0, stream>>>(qkv_w,  qkvw_bf, QKV_D * C_DIM);
  f32_to_bf16_kernel<<<2048, 256, 0, stream>>>(proj_w, projw_bf, C_DIM * N_HEAD * HS);
  f32_to_bf16_kernel<<<2048, 256, 0, stream>>>(fc1_w,  fc1w_bf, FFN * C_DIM);
  f32_to_bf16_kernel<<<2048, 256, 0, stream>>>(fc2_w,  fc2w_bf, FFN * C_DIM);
  f32_to_bf16_kernel<<<2048, 256, 0, stream>>>(mlp_w,  mlpw_bf, C_DIM * FFN);

  // 2. rmsnorm1
  rmsnorm_kernel<<<T_SEQ, 256, 0, stream>>>(x, norm1_w, xn_bf);

  // 3. qkv = xn @ qkv_w^T   (2048 x 4096, K=2048)
  gemm_nt_bf16_kernel<<<dim3(QKV_D / 128, T_SEQ / 128), 256, 0, stream>>>(
      xn_bf, qkvw_bf, qkv_f, T_SEQ, QKV_D, C_DIM);

  // 4. rope + split
  rope_split_kernel<<<T_SEQ, 256, 0, stream>>>(qkv_f, cosv, sinv, q_bf, k_bf, vt_bf);

  // 5. flash attention
  flash_attn_kernel<<<dim3(T_SEQ / 128, N_HEAD), 256, 0, stream>>>(q_bf, k_bf, vt_bf, y_bf);

  // 6. attn proj
  gemm_nt_bf16_kernel<<<dim3(C_DIM / 128, T_SEQ / 128), 256, 0, stream>>>(
      y_bf, projw_bf, attnp_f, T_SEQ, C_DIM, N_HEAD * HS);

  // 7. residual
  add_f32_kernel<<<(T_SEQ * C_DIM + 255) / 256, 256, 0, stream>>>(
      attnp_f, x, x2_f, T_SEQ * C_DIM);

  // 8. rmsnorm2
  rmsnorm_kernel<<<T_SEQ, 256, 0, stream>>>(x2_f, norm2_w, xn2_bf);

  // 9/10. fc1, fc2
  gemm_nt_bf16_kernel<<<dim3(FFN / 128, T_SEQ / 128), 256, 0, stream>>>(
      xn2_bf, fc1w_bf, h1_f, T_SEQ, FFN, C_DIM);
  gemm_nt_bf16_kernel<<<dim3(FFN / 128, T_SEQ / 128), 256, 0, stream>>>(
      xn2_bf, fc2w_bf, h2_f, T_SEQ, FFN, C_DIM);

  // 11. silu * gate
  silu_mul_kernel<<<4096, 256, 0, stream>>>(h1_f, h2_f, h_bf, T_SEQ * FFN);

  // 12. mlp proj (K=5632)
  gemm_nt_bf16_kernel<<<dim3(C_DIM / 128, T_SEQ / 128), 256, 0, stream>>>(
      h_bf, mlpw_bf, mlp_f, T_SEQ, C_DIM, FFN);

  // 13. final residual -> output
  add_f32_kernel<<<(T_SEQ * C_DIM + 255) / 256, 256, 0, stream>>>(
      mlp_f, x2_f, out, T_SEQ * C_DIM);
}
